// FluxSingleAttention_63273458205329
// MI455X (gfx1250) — compile-verified, matrix operand order checked by reading, not run on hardware
//
#include <hip/hip_runtime.h>

#define NUM_HEADS 24
#define HEAD_DIM  128
#define DIM       3072
#define N3        (3 * DIM)        // 9216
#define BATCH     2
#define SEQ       2048
#define MROWS     (BATCH * SEQ)    // 4096
#define SCALE     0.08838834764831845f  // 1/sqrt(128)

typedef __bf16 bf16_t;
typedef bf16_t v16bf __attribute__((ext_vector_type(16)));
typedef bf16_t v8bf  __attribute__((ext_vector_type(8)));
typedef float  v8f   __attribute__((ext_vector_type(8)));

// ---------------------------------------------------------------------------
// Fragment load: 16x32 bf16 A-layout (also used for B fragments of a matrix
// stored so the WMMA K-dim is contiguous).  wave32 ISA layout:
//   lane = m + 16*kh ; elements 0..7 -> K = k0+kh*8+0..7 ;
//                      elements 8..15 -> K = k0+16+kh*8+0..7
// Two contiguous 16B loads per lane (global_load_b128 or ds_load_b128).
// ---------------------------------------------------------------------------
__device__ __forceinline__ v16bf load_frag(const bf16_t* base, int ld,
                                           int row0, int k0, int lane) {
  const int m  = lane & 15;
  const int kh = (lane >> 4) & 1;
  const bf16_t* p = base + (size_t)(row0 + m) * (size_t)ld + (size_t)(k0 + (kh << 3));
  v8bf lo = *(const v8bf*)(p);
  v8bf hi = *(const v8bf*)(p + 16);
  return __builtin_shufflevector(lo, hi, 0,1,2,3,4,5,6,7,8,9,10,11,12,13,14,15);
}

__device__ __forceinline__ v8f wmma_bf16(v16bf a, v16bf b, v8f c) {
  return __builtin_amdgcn_wmma_f32_16x16x32_bf16(false, a, false, b,
                                                 (short)0, c, false, false);
}

// ---------------------------------------------------------------------------
// CDNA5 async global->LDS copy (ASYNCcnt-tracked DMA, 16B per lane).
// vdst = per-lane LDS byte offset; vaddr = per-lane 64-bit global address.
// ---------------------------------------------------------------------------
__device__ __forceinline__ void async_copy_b128(bf16_t* lds_dst, const bf16_t* gsrc) {
  asm volatile("global_load_async_to_lds_b128 %0, %1, off"
               :
               : "v"((unsigned)(uintptr_t)lds_dst), "v"(gsrc)
               : "memory");
}

__device__ __forceinline__ void wait_async0() {
#if __has_builtin(__builtin_amdgcn_s_wait_asynccnt)
  __builtin_amdgcn_s_wait_asynccnt(0);
#else
  asm volatile("s_wait_asynccnt 0x0" ::: "memory");
#endif
}

// ---------------------------------------------------------------------------
// Kernel 1a: f32 -> bf16 copy of hidden_states
// ---------------------------------------------------------------------------
__global__ __launch_bounds__(256) void convert_x(const float* __restrict__ x,
                                                 bf16_t* __restrict__ xb, size_t n) {
  for (size_t i = (size_t)blockIdx.x * blockDim.x + threadIdx.x; i < n;
       i += (size_t)gridDim.x * blockDim.x)
    xb[i] = (bf16_t)x[i];
}

// ---------------------------------------------------------------------------
// Kernel 1b: W[K=3072][N=9216] f32 -> Wt[N][K] bf16 (WMMA B frags K-contig).
// ---------------------------------------------------------------------------
__global__ __launch_bounds__(256) void transpose_w(const float* __restrict__ w,
                                                   bf16_t* __restrict__ wt) {
  __shared__ float tile[32][33];
  const int n0 = blockIdx.x * 32;
  const int k0 = blockIdx.y * 32;
  const int tx = threadIdx.x & 31;
  const int ty = threadIdx.x >> 5;  // 0..7
#pragma unroll
  for (int i = ty; i < 32; i += 8)
    tile[i][tx] = w[(size_t)(k0 + i) * N3 + n0 + tx];
  __syncthreads();
#pragma unroll
  for (int i = ty; i < 32; i += 8)
    wt[(size_t)(n0 + i) * DIM + k0 + tx] = (bf16_t)tile[tx][i];
}

// ---------------------------------------------------------------------------
// Kernel 2: QKV GEMM, bf16 WMMA, f32 accumulate + bias.
// Block = 256 thr = 8 waves (2x4).  Block tile 64x256, wave tile 32x64.
// A/B tiles double-buffered in LDS via async global->LDS DMA (ASYNCcnt),
// issued for tile k+1 before computing tile k; s_wait_asynccnt 0 + barrier
// flips buffers.  LDS rows padded 32->40 elems (80B stride) so the 16-lane
// b128 fragment reads spread over the 64 banks.
// ---------------------------------------------------------------------------
#define KC   32
#define LDP  40   // padded LDS row length (elements)

__global__ __launch_bounds__(256) void qkv_gemm(const bf16_t* __restrict__ X,
                                                const bf16_t* __restrict__ Wt,
                                                const float* __restrict__ bias,
                                                float* __restrict__ out) {
  __shared__ __align__(16) bf16_t As[2][64][LDP];    //  2 x 5 KB
  __shared__ __align__(16) bf16_t Bs[2][256][LDP];   //  2 x 20 KB

  const int tid  = threadIdx.x;
  const int lane = tid & 31;
  const int wave = tid >> 5;
  const int bm = blockIdx.y * 64;
  const int bn = blockIdx.x * 256;
  const int wml = (wave >> 2) * 32;   // wave tile origin inside block tile
  const int wnl = (wave & 3) * 64;

  // stage one 64x32 A tile + 256x32 B tile into LDS buffer `buf`
  auto stage = [&](int buf, int k0) {
    const int arow = tid >> 2, ac = (tid & 3) << 3;         // 256 x b128
    async_copy_b128(&As[buf][arow][ac],
                    X + (size_t)(bm + arow) * DIM + k0 + ac);
#pragma unroll
    for (int i = 0; i < 4; ++i) {                            // 1024 x b128
      const int idx = tid + (i << 8);
      const int brow = idx >> 2, bc = (idx & 3) << 3;
      async_copy_b128(&Bs[buf][brow][bc],
                      Wt + (size_t)(bn + brow) * DIM + k0 + bc);
    }
  };

  stage(0, 0);
  wait_async0();
  __syncthreads();

  v8f acc[2][4] = {};
  for (int k0 = 0; k0 < DIM; k0 += KC) {
    const int buf = (k0 >> 5) & 1;
    if (k0 + KC < DIM) stage(buf ^ 1, k0 + KC);   // prefetch next tile (DMA)

    v16bf a0 = load_frag(&As[buf][0][0], LDP, wml, 0, lane);
    v16bf a1 = load_frag(&As[buf][0][0], LDP, wml + 16, 0, lane);
    v16bf b[4];
#pragma unroll
    for (int j = 0; j < 4; ++j)
      b[j] = load_frag(&Bs[buf][0][0], LDP, wnl + j * 16, 0, lane);
#pragma unroll
    for (int j = 0; j < 4; ++j) {
      acc[0][j] = wmma_bf16(a0, b[j], acc[0][j]);
      acc[1][j] = wmma_bf16(a1, b[j], acc[1][j]);
    }

    wait_async0();     // our share of next-tile DMA done
    __syncthreads();   // everyone done reading `buf` / writing `buf^1`
  }

  // epilogue: C tile lane map n = lane&15, rows r + 8*(lane>>4)
  const int n = lane & 15;
  const int hh = lane >> 4;
#pragma unroll
  for (int j = 0; j < 4; ++j) {
    const int col = bn + wnl + j * 16 + n;
    const float bv = bias[col];
#pragma unroll
    for (int i = 0; i < 2; ++i)
#pragma unroll
      for (int r = 0; r < 8; ++r)
        out[(size_t)(bm + wml + i * 16 + r + 8 * hh) * N3 + col] = acc[i][j][r] + bv;
  }
}

// ---------------------------------------------------------------------------
// Kernel 3: fused RMSNorm (Q,K) + RoPE (Q,K) + pack to bf16.
//   qb, kb : [B, H, S, D]   (tokens-major, D contiguous -> frag-friendly)
//   vt     : [B, H, D, S]   (V transposed: keys contiguous per d row)
// ---------------------------------------------------------------------------
__global__ __launch_bounds__(128) void norm_rope_pack(
    const float* __restrict__ qkvf, const float* __restrict__ rot,
    const float* __restrict__ qw, const float* __restrict__ kw,
    bf16_t* __restrict__ qb, bf16_t* __restrict__ kb, bf16_t* __restrict__ vt) {
  const int s = blockIdx.x, h = blockIdx.y, b = blockIdx.z;
  const int d = threadIdx.x;

  __shared__ float bufq[128], bufk[128], rq[128], rk[128];

  const size_t row = ((size_t)b * SEQ + s) * N3;
  const float xq = qkvf[row + 0 * DIM + h * HEAD_DIM + d];
  const float xk = qkvf[row + 1 * DIM + h * HEAD_DIM + d];
  const float xv = qkvf[row + 2 * DIM + h * HEAD_DIM + d];

  bufq[d] = xq;  bufk[d] = xk;
  rq[d] = xq * xq;  rk[d] = xk * xk;
  __syncthreads();
#pragma unroll
  for (int off = 64; off; off >>= 1) {
    if (d < off) { rq[d] += rq[d + off]; rk[d] += rk[d + off]; }
    __syncthreads();
  }
  const float invq = rsqrtf(rq[0] * (1.0f / HEAD_DIM) + 1e-6f);
  const float invk = rsqrtf(rk[0] * (1.0f / HEAD_DIM) + 1e-6f);

  const float* fr = rot + ((size_t)s * (HEAD_DIM / 2) + (d >> 1)) * 4;
  const float f0 = fr[(d & 1) * 2 + 0];
  const float f1 = fr[(d & 1) * 2 + 1];

  const float q0 = bufq[d & ~1] * invq * qw[d & ~1];
  const float q1 = bufq[d |  1] * invq * qw[d |  1];
  const float k0 = bufk[d & ~1] * invk * kw[d & ~1];
  const float k1 = bufk[d |  1] * invk * kw[d |  1];

  const size_t bh = (size_t)(b * NUM_HEADS + h);
  qb[(bh * SEQ + s) * HEAD_DIM + d] = (bf16_t)(f0 * q0 + f1 * q1);
  kb[(bh * SEQ + s) * HEAD_DIM + d] = (bf16_t)(f0 * k0 + f1 * k1);
  vt[(bh * HEAD_DIM + d) * SEQ + s] = (bf16_t)xv;
}

// ---------------------------------------------------------------------------
// Kernel 4: flash attention. Block = 256 thr = 8 waves; each wave owns 16
// query rows.  Per 32-key step: 8 WMMA (QK^T), online softmax with 16-lane
// shfl_xor reductions, P restaged via wave-private LDS into A-fragment
// layout, 8 WMMA (P*V against V^T).
// ---------------------------------------------------------------------------
__global__ __launch_bounds__(256) void flash_attn(const bf16_t* __restrict__ qb,
                                                  const bf16_t* __restrict__ kb,
                                                  const bf16_t* __restrict__ vt,
                                                  float* __restrict__ out) {
  const int lane = threadIdx.x & 31;
  const int wave = threadIdx.x >> 5;
  const int b = blockIdx.z, h = blockIdx.y;
  const int q0 = blockIdx.x * 128 + wave * 16;

  const size_t bh = (size_t)(b * NUM_HEADS + h);
  const bf16_t* Q  = qb + bh * SEQ * HEAD_DIM;
  const bf16_t* Km = kb + bh * SEQ * HEAD_DIM;
  const bf16_t* Vt = vt + bh * HEAD_DIM * SEQ;

  __shared__ __align__(16) bf16_t pbuf[8][16][32];  // per-wave P staging

  v16bf qf[4];
#pragma unroll
  for (int c = 0; c < 4; ++c) qf[c] = load_frag(Q, HEAD_DIM, q0, c * 32, lane);

  v8f o[8] = {};
  float rmax[8], rsum[8];
#pragma unroll
  for (int r = 0; r < 8; ++r) { rmax[r] = -1e30f; rsum[r] = 0.0f; }

  for (int kv = 0; kv < SEQ; kv += 32) {
    v8f s0 = {}, s1 = {};
#pragma unroll
    for (int c = 0; c < 4; ++c) {
      v16bf kf0 = load_frag(Km, HEAD_DIM, kv,      c * 32, lane);
      v16bf kf1 = load_frag(Km, HEAD_DIM, kv + 16, c * 32, lane);
      s0 = wmma_bf16(qf[c], kf0, s0);
      s1 = wmma_bf16(qf[c], kf1, s1);
    }
#pragma unroll
    for (int r = 0; r < 8; ++r) {
      float a = s0[r] * SCALE;
      float bsc = s1[r] * SCALE;
      float t = fmaxf(a, bsc);
#pragma unroll
      for (int off = 1; off < 16; off <<= 1) t = fmaxf(t, __shfl_xor(t, off, 32));
      const float nm = fmaxf(rmax[r], t);
      const float al = __expf(rmax[r] - nm);
      rmax[r] = nm;
      const float p0 = __expf(a - nm);
      const float p1 = __expf(bsc - nm);
      float ps = p0 + p1;
#pragma unroll
      for (int off = 1; off < 16; off <<= 1) ps += __shfl_xor(ps, off, 32);
      rsum[r] = rsum[r] * al + ps;
#pragma unroll
      for (int t8 = 0; t8 < 8; ++t8) o[t8][r] *= al;
      s0[r] = p0;
      s1[r] = p1;
    }
    {
      const int n = lane & 15, hh = lane >> 4;
#pragma unroll
      for (int r = 0; r < 8; ++r) {
        pbuf[wave][r + 8 * hh][n]      = (bf16_t)s0[r];
        pbuf[wave][r + 8 * hh][16 + n] = (bf16_t)s1[r];
      }
    }
    v16bf pf = load_frag(&pbuf[wave][0][0], 32, 0, 0, lane);
#pragma unroll
    for (int t8 = 0; t8 < 8; ++t8) {
      v16bf vf = load_frag(Vt, SEQ, t8 * 16, kv, lane);
      o[t8] = wmma_bf16(pf, vf, o[t8]);
    }
  }

  const int n = lane & 15, hh = lane >> 4;
#pragma unroll
  for (int t8 = 0; t8 < 8; ++t8)
#pragma unroll
    for (int r = 0; r < 8; ++r) {
      const int m = r + 8 * hh;
      out[((size_t)b * SEQ + (q0 + m)) * (NUM_HEADS * HEAD_DIM)
          + h * HEAD_DIM + t8 * 16 + n] = o[t8][r] / rsum[r];
    }
}

// ---------------------------------------------------------------------------
extern "C" void kernel_launch(void* const* d_in, const int* in_sizes, int n_in,
                              void* d_out, int out_size, void* d_ws, size_t ws_size,
                              hipStream_t stream) {
  (void)in_sizes; (void)n_in; (void)out_size; (void)ws_size;
  const float* hs   = (const float*)d_in[0];   // [B,S,3072]
  const float* rot  = (const float*)d_in[1];   // [1,1,S,64,2,2]
  const float* wqkv = (const float*)d_in[2];   // [3072,9216]
  const float* bqkv = (const float*)d_in[3];   // [9216]
  const float* qw   = (const float*)d_in[4];   // [128]
  const float* kw   = (const float*)d_in[5];   // [128]
  float* out = (float*)d_out;                  // [B,S,3072]

  char* p = (char*)d_ws;
  bf16_t* Xb  = (bf16_t*)p;  p += (size_t)MROWS * DIM * sizeof(bf16_t);     //  25 MB
  bf16_t* Wt  = (bf16_t*)p;  p += (size_t)N3 * DIM * sizeof(bf16_t);       //  57 MB
  float*  qkv = (float*)p;   p += (size_t)MROWS * N3 * sizeof(float);      // 151 MB
  const size_t hsd = (size_t)BATCH * NUM_HEADS * SEQ * HEAD_DIM;
  bf16_t* qb = (bf16_t*)p;   p += hsd * sizeof(bf16_t);                    //  25 MB
  bf16_t* kb = (bf16_t*)p;   p += hsd * sizeof(bf16_t);                    //  25 MB
  bf16_t* vt = (bf16_t*)p;                                                 //  25 MB

  convert_x<<<4096, 256, 0, stream>>>(hs, Xb, (size_t)MROWS * DIM);
  transpose_w<<<dim3(N3 / 32, DIM / 32), 256, 0, stream>>>(wqkv, Wt);
  qkv_gemm<<<dim3(N3 / 256, MROWS / 64), 256, 0, stream>>>(Xb, Wt, bqkv, qkv);
  norm_rope_pack<<<dim3(SEQ, NUM_HEADS, BATCH), 128, 0, stream>>>(
      qkv, rot, qw, kw, qb, kb, vt);
  flash_attn<<<dim3(SEQ / 128, NUM_HEADS, BATCH), 256, 0, stream>>>(
      qb, kb, vt, out);
}
